// SpatioEmbeddedRNN_21930103013499
// MI455X (gfx1250) — matrix operand order repaired and checked by reading, not compiled, not run on hardware
//
#include <hip/hip_runtime.h>
#include <math.h>

// ---------------------------------------------------------------------------
// Problem constants
// ---------------------------------------------------------------------------
#define BB   32
#define TT   24
#define NN   256
#define FF   16
#define HH   128
#define H2   64
#define OUTD 24
#define FH   144        // F + H
#define FHP  160        // F + H padded to multiple of 32 (WMMA K-chunk)
#define G4   512        // 4*H gates

typedef _Float16 v16h __attribute__((ext_vector_type(16)));
typedef _Float16 v8h  __attribute__((ext_vector_type(8)));
typedef float    v8f  __attribute__((ext_vector_type(8)));

union FragU { v16h v; v8h h[2]; };

__device__ __forceinline__ float sigf(float x) { return 1.0f / (1.0f + expf(-x)); }

// ---------------------------------------------------------------------------
// Init kernels
// ---------------------------------------------------------------------------
__global__ void conv_adj_kernel(const float* __restrict__ adj, _Float16* __restrict__ Af16) {
    int idx = blockIdx.x * blockDim.x + threadIdx.x;          // B*N*N = 2M
    if (idx >= BB * NN * NN) return;
    int b    = idx >> 16;                                      // /65536
    int rest = idx & 65535;
    Af16[idx] = (_Float16)adj[((size_t)b * TT + (TT - 1)) * NN * NN + rest];
}

__global__ void build_WT_kernel(const float* __restrict__ gcn_W, _Float16* __restrict__ WT) {
    int idx = blockIdx.x * blockDim.x + threadIdx.x;          // 512*160
    if (idx >= G4 * FHP) return;
    int o = idx / FHP;
    int k = idx % FHP;
    WT[idx] = (k < FH) ? (_Float16)gcn_W[(size_t)k * G4 + o] : (_Float16)0.0f;
}

__global__ void zero_f16_kernel(_Float16* __restrict__ p, int n) {
    int idx = blockIdx.x * blockDim.x + threadIdx.x;
    if (idx < n) p[idx] = (_Float16)0.0f;
}

__global__ void zero_f32_kernel(float* __restrict__ p, int n) {
    int idx = blockIdx.x * blockDim.x + threadIdx.x;
    if (idx < n) p[idx] = 0.0f;
}

// Write x_t into the transposed xh buffer: xhT[b][f][node], f in [0,16)
__global__ void set_xpart_kernel(const float* __restrict__ x, _Float16* __restrict__ xhT, int t) {
    int idx = blockIdx.x * blockDim.x + threadIdx.x;          // B*N*F = 131072
    if (idx >= BB * NN * FF) return;
    int f    = idx & 15;
    int node = (idx >> 4) & 255;
    int b    = idx >> 12;
    xhT[(size_t)b * FH * NN + (size_t)f * NN + node] =
        (_Float16)x[(((size_t)b * TT + t) * NN + node) * FF + f];
}

// ---------------------------------------------------------------------------
// Fragment loaders (CDNA5 wave32 WMMA layouts)
//   A 16x32 f16 : lane l (g=l/16, m=l%16) holds K = {8g+0..7, 16+8g+0..7}
//   B 32x16 f16 : lane l holds col n=l%16,  K = {8g+0..7, 16+8g+0..7}
// ---------------------------------------------------------------------------
__device__ __forceinline__ v16h load_fragA(const _Float16* __restrict__ rowPtr, int g) {
    FragU a;
    a.h[0] = *(const v8h*)(rowPtr + 8 * g);
    a.h[1] = *(const v8h*)(rowPtr + 8 * g + 16);
    return a.v;
}
__device__ __forceinline__ v16h load_fragB(const _Float16* __restrict__ colPtr, int g) {
    FragU b;
    b.h[0] = *(const v8h*)(colPtr + 8 * g);
    b.h[1] = *(const v8h*)(colPtr + 8 * g + 16);
    return b.v;
}

// ---------------------------------------------------------------------------
// WMMA kernel 1: agg[b] = A[b] (256x256) @ xh[b] (256x144), f32 acc -> f16 out
// Each wave: 1 M-tile x 3 N-tiles (A fragment reused 3x). 4 waves / block.
// grid = (16 M-tiles * 3 N-groups / 4 = 12, B)
// ---------------------------------------------------------------------------
__global__ __launch_bounds__(128) void gcn_agg_kernel(const _Float16* __restrict__ Af16,
                                                      const _Float16* __restrict__ xhT,
                                                      _Float16* __restrict__ agg) {
    const int b    = blockIdx.y;
    const int wid  = blockIdx.x * 4 + (threadIdx.x >> 5);      // 0..47
    const int m0   = (wid / 3) * 16;
    const int n0   = (wid % 3) * 48;                           // 3 sub-tiles of 16
    const int lane = threadIdx.x & 31;
    const int g    = lane >> 4;
    const int ln   = lane & 15;

    const _Float16* Ab = Af16 + (size_t)b * NN * NN + (size_t)(m0 + ln) * NN;
    const _Float16* Xb = xhT  + (size_t)b * FH * NN;

    v8f acc[3] = {};
    #pragma unroll
    for (int k0 = 0; k0 < NN; k0 += 32) {
        v16h a = load_fragA(Ab + k0, g);
        #pragma unroll
        for (int s = 0; s < 3; ++s) {
            v16h bf = load_fragB(Xb + (size_t)(n0 + 16 * s + ln) * NN + k0, g);
            acc[s] = __builtin_amdgcn_wmma_f32_16x16x32_f16(false, a, false, bf,
                                                            (short)0, acc[s], false, false);
        }
    }

    _Float16* ag = agg + (size_t)b * NN * FHP;
    #pragma unroll
    for (int s = 0; s < 3; ++s) {
        #pragma unroll
        for (int r = 0; r < 8; ++r) {
            ag[(size_t)(m0 + r + 8 * g) * FHP + n0 + 16 * s + ln] = (_Float16)acc[s][r];
        }
    }
}

// ---------------------------------------------------------------------------
// WMMA kernel 2: gates[b] = agg[b] (256x144, K padded to 160) @ gcn_W (144x512)
// Each wave: 1 M-tile x 4 N-tiles. 4 waves / block.
// grid = (16 M-tiles * 8 N-groups / 4 = 32, B)
// ---------------------------------------------------------------------------
__global__ __launch_bounds__(128) void gcn_gates_kernel(const _Float16* __restrict__ agg,
                                                        const _Float16* __restrict__ WT,
                                                        float* __restrict__ gates) {
    const int b    = blockIdx.y;
    const int wid  = blockIdx.x * 4 + (threadIdx.x >> 5);      // 0..127
    const int m0   = (wid >> 3) * 16;
    const int n0   = (wid & 7) * 64;                           // 4 sub-tiles of 16
    const int lane = threadIdx.x & 31;
    const int g    = lane >> 4;
    const int ln   = lane & 15;

    const _Float16* Ab = agg + (size_t)b * NN * FHP + (size_t)(m0 + ln) * FHP;

    v8f acc[4] = {};
    #pragma unroll
    for (int k0 = 0; k0 < FHP; k0 += 32) {
        v16h a = load_fragA(Ab + k0, g);
        #pragma unroll
        for (int s = 0; s < 4; ++s) {
            v16h bf = load_fragB(WT + (size_t)(n0 + 16 * s + ln) * FHP + k0, g);
            acc[s] = __builtin_amdgcn_wmma_f32_16x16x32_f16(false, a, false, bf,
                                                            (short)0, acc[s], false, false);
        }
    }

    float* gp = gates + (size_t)b * NN * G4;
    #pragma unroll
    for (int s = 0; s < 4; ++s) {
        #pragma unroll
        for (int r = 0; r < 8; ++r) {
            gp[(size_t)(m0 + r + 8 * g) * G4 + n0 + 16 * s + ln] = acc[s][r];
        }
    }
}

// ---------------------------------------------------------------------------
// LSTM elementwise update; writes new h into f32 state AND transposed f16 xhT
// ---------------------------------------------------------------------------
__global__ void lstm_update_kernel(const float* __restrict__ gates,
                                   const float* __restrict__ gcn_b,
                                   float* __restrict__ c, float* __restrict__ h,
                                   _Float16* __restrict__ xhT) {
    int idx = blockIdx.x * blockDim.x + threadIdx.x;          // B*N*H = 1M
    if (idx >= BB * NN * HH) return;
    int hc   = idx & 127;
    int node = (idx >> 7) & 255;
    int b    = idx >> 15;

    const float* gr = gates + ((size_t)b * NN + node) * G4;
    float gi = gr[hc]          + gcn_b[hc];
    float gf = gr[HH + hc]     + gcn_b[HH + hc];
    float gg = gr[2 * HH + hc] + gcn_b[2 * HH + hc];
    float go = gr[3 * HH + hc] + gcn_b[3 * HH + hc];

    float cv = sigf(gf) * c[idx] + sigf(gi) * tanhf(gg);
    float hv = sigf(go) * tanhf(cv);
    c[idx] = cv;
    h[idx] = hv;
    xhT[(size_t)b * FH * NN + (size_t)(FF + hc) * NN + node] = (_Float16)hv;
}

// ---------------------------------------------------------------------------
// x_spatial = relu(h[:, station]) @ fc5_W.T + fc5_b   (32 x 128) — tiny
// ---------------------------------------------------------------------------
__global__ void spatial_fc5_kernel(const float* __restrict__ h,
                                   const float* __restrict__ fc5W,
                                   const float* __restrict__ fc5b,
                                   const int* __restrict__ tsi,
                                   float* __restrict__ xsp) {
    int b = blockIdx.x;
    int o = threadIdx.x;                                       // 128
    __shared__ float hs[HH];
    int st = *tsi;
    hs[o] = fmaxf(h[((size_t)b * NN + st) * HH + o], 0.0f);
    __syncthreads();
    float a = fc5b[o];
    for (int k = 0; k < HH; ++k) a += fc5W[(size_t)o * HH + k] * hs[k];
    xsp[(size_t)b * HH + o] = a;
}

// ---------------------------------------------------------------------------
// GRU over the target-station time series — tiny, one block per batch
// ---------------------------------------------------------------------------
__global__ void gru_kernel(const float* __restrict__ x, const int* __restrict__ tsi,
                           const float* __restrict__ Wih, const float* __restrict__ Whh,
                           const float* __restrict__ bih, const float* __restrict__ bhh,
                           float* __restrict__ hgru) {
    int b = blockIdx.x;
    int j = threadIdx.x;                                       // 128
    __shared__ float hs[HH];
    __shared__ float xs[FF];
    hs[j] = 0.0f;
    int st = *tsi;
    for (int t = 0; t < TT; ++t) {
        __syncthreads();
        if (j < FF) xs[j] = x[(((size_t)b * TT + t) * NN + st) * FF + j];
        __syncthreads();
        float gr = bih[j], gz = bih[HH + j], gn = bih[2 * HH + j];
        for (int f = 0; f < FF; ++f) {
            float xv = xs[f];
            gr += Wih[(size_t)j * FF + f] * xv;
            gz += Wih[(size_t)(HH + j) * FF + f] * xv;
            gn += Wih[(size_t)(2 * HH + j) * FF + f] * xv;
        }
        float hr = bhh[j], hz = bhh[HH + j], hn = bhh[2 * HH + j];
        for (int k = 0; k < HH; ++k) {
            float hv = hs[k];
            hr += Whh[(size_t)j * HH + k] * hv;
            hz += Whh[(size_t)(HH + j) * HH + k] * hv;
            hn += Whh[(size_t)(2 * HH + j) * HH + k] * hv;
        }
        float r = sigf(gr + hr);
        float z = sigf(gz + hz);
        float n = tanhf(gn + r * hn);
        float hnew = (1.0f - z) * n + z * hs[j];
        __syncthreads();
        hs[j] = hnew;
    }
    hgru[(size_t)b * HH + j] = hs[j];
}

// ---------------------------------------------------------------------------
// Head: fc2 (256->128, relu) -> fc3 (128->64, relu) -> fc4 (64->24)
// ---------------------------------------------------------------------------
__global__ void head_kernel(const float* __restrict__ hgru, const float* __restrict__ xsp,
                            const float* __restrict__ fc2W, const float* __restrict__ fc2b,
                            const float* __restrict__ fc3W, const float* __restrict__ fc3b,
                            const float* __restrict__ fc4W, const float* __restrict__ fc4b,
                            float* __restrict__ out) {
    int b = blockIdx.x;
    int j = threadIdx.x;                                       // 256
    __shared__ float hc[2 * HH];
    __shared__ float y2[HH];
    __shared__ float y3[H2];
    hc[j] = (j < HH) ? hgru[(size_t)b * HH + j] : xsp[(size_t)b * HH + (j - HH)];
    __syncthreads();
    if (j < HH) {
        float a = fc2b[j];
        for (int k = 0; k < 2 * HH; ++k) a += fc2W[(size_t)j * 2 * HH + k] * hc[k];
        y2[j] = fmaxf(a, 0.0f);
    }
    __syncthreads();
    if (j < H2) {
        float a = fc3b[j];
        for (int k = 0; k < HH; ++k) a += fc3W[(size_t)j * HH + k] * y2[k];
        y3[j] = fmaxf(a, 0.0f);
    }
    __syncthreads();
    if (j < OUTD) {
        float a = fc4b[j];
        for (int k = 0; k < H2; ++k) a += fc4W[(size_t)j * H2 + k] * y3[k];
        out[(size_t)b * OUTD + j] = a;
    }
}

// ---------------------------------------------------------------------------
// Host launcher
// ---------------------------------------------------------------------------
extern "C" void kernel_launch(void* const* d_in, const int* in_sizes, int n_in,
                              void* d_out, int out_size, void* d_ws, size_t ws_size,
                              hipStream_t stream) {
    (void)in_sizes; (void)n_in; (void)out_size; (void)ws_size;
    const float* x     = (const float*)d_in[0];
    const float* adj   = (const float*)d_in[1];
    const int*   tsi   = (const int*)  d_in[2];
    const float* gcn_W = (const float*)d_in[3];
    const float* gcn_b = (const float*)d_in[4];
    const float* gWih  = (const float*)d_in[5];
    const float* gWhh  = (const float*)d_in[6];
    const float* gbih  = (const float*)d_in[7];
    const float* gbhh  = (const float*)d_in[8];
    const float* fc2W  = (const float*)d_in[9];
    const float* fc2b  = (const float*)d_in[10];
    const float* fc3W  = (const float*)d_in[11];
    const float* fc3b  = (const float*)d_in[12];
    const float* fc4W  = (const float*)d_in[13];
    const float* fc4b  = (const float*)d_in[14];
    const float* fc5W  = (const float*)d_in[15];
    const float* fc5b  = (const float*)d_in[16];
    float* out = (float*)d_out;

    char* ws = (char*)d_ws;
    size_t off = 0;
    auto alloc = [&](size_t bytes) -> void* {
        void* p = ws + off;
        off += (bytes + 255) & ~(size_t)255;
        return p;
    };
    _Float16* Af16 = (_Float16*)alloc((size_t)BB * NN * NN * 2);   // 4 MB
    _Float16* WT   = (_Float16*)alloc((size_t)G4 * FHP * 2);       // 160 KB
    _Float16* xhT  = (_Float16*)alloc((size_t)BB * FH * NN * 2);   // 2.25 MB
    _Float16* agg  = (_Float16*)alloc((size_t)BB * NN * FHP * 2);  // 2.5 MB
    float*    gates= (float*)   alloc((size_t)BB * NN * G4 * 4);   // 16 MB
    float*    hbuf = (float*)   alloc((size_t)BB * NN * HH * 4);   // 4 MB
    float*    cbuf = (float*)   alloc((size_t)BB * NN * HH * 4);   // 4 MB
    float*    xsp  = (float*)   alloc((size_t)BB * HH * 4);
    float*    hgru = (float*)   alloc((size_t)BB * HH * 4);

    const int TPB = 256;
    // ---- init (every call: deterministic, ws is not re-poisoned) ----
    conv_adj_kernel<<<(BB * NN * NN + TPB - 1) / TPB, TPB, 0, stream>>>(adj, Af16);
    build_WT_kernel<<<(G4 * FHP + TPB - 1) / TPB, TPB, 0, stream>>>(gcn_W, WT);
    zero_f16_kernel<<<(BB * FH * NN + TPB - 1) / TPB, TPB, 0, stream>>>(xhT, BB * FH * NN);
    zero_f16_kernel<<<(BB * NN * FHP + TPB - 1) / TPB, TPB, 0, stream>>>(agg, BB * NN * FHP);
    zero_f32_kernel<<<(BB * NN * HH + TPB - 1) / TPB, TPB, 0, stream>>>(cbuf, BB * NN * HH);

    // ---- GCN-LSTM scan over T steps ----
    dim3 gAgg(12, BB);     // 48 wave-tiles / 4 waves per block
    dim3 gGts(32, BB);     // 128 wave-tiles / 4 waves per block
    for (int t = 0; t < TT; ++t) {
        set_xpart_kernel<<<(BB * NN * FF + TPB - 1) / TPB, TPB, 0, stream>>>(x, xhT, t);
        gcn_agg_kernel  <<<gAgg, 128, 0, stream>>>(Af16, xhT, agg);
        gcn_gates_kernel<<<gGts, 128, 0, stream>>>(agg, WT, gates);
        lstm_update_kernel<<<(BB * NN * HH + TPB - 1) / TPB, TPB, 0, stream>>>(
            gates, gcn_b, cbuf, hbuf, xhT);
    }

    // ---- small tail ----
    spatial_fc5_kernel<<<BB, HH, 0, stream>>>(hbuf, fc5W, fc5b, tsi, xsp);
    gru_kernel<<<BB, HH, 0, stream>>>(x, tsi, gWih, gWhh, gbih, gbhh, hgru);
    head_kernel<<<BB, 2 * HH, 0, stream>>>(hgru, xsp, fc2W, fc2b, fc3W, fc3b,
                                           fc4W, fc4b, out);
}
